// TransformerBlock_12524124635845
// MI455X (gfx1250) — matrix operand order repaired and checked by reading, not compiled, not run on hardware
//
#include <hip/hip_runtime.h>
#include <hip/hip_bf16.h>

// ---------------------------------------------------------------------------
// Problem constants (transformer block, eval mode)
// ---------------------------------------------------------------------------
#define EMBED 1024
#define FFDIM 4096
#define SEQ   2048
#define BATCH 4
#define NHEAD 16
#define HD    64
#define MTOT  (BATCH*SEQ)          // 8192 token rows

typedef __attribute__((ext_vector_type(16))) __bf16 v16bf;
typedef __attribute__((ext_vector_type(8)))  float  v8f;

union Frag { uint4 q[2]; v16bf bf; };   // 32 bytes: one WMMA A/B fragment per lane

__device__ __forceinline__ unsigned short f2bf(float f) {
  unsigned int u = __builtin_bit_cast(unsigned int, f);
  u += 0x7fffu + ((u >> 16) & 1u);      // round-to-nearest-even
  return (unsigned short)(u >> 16);
}

__device__ __forceinline__ v8f wmma_bf16(const Frag& a, const Frag& b, v8f c) {
  return __builtin_amdgcn_wmma_f32_16x16x32_bf16(
      /*neg_a=*/false, a.bf, /*neg_b=*/false, b.bf,
      /*c_mod=*/(short)0, c, /*reuse_a=*/false, /*reuse_b=*/false);
}

// ---------------------------------------------------------------------------
// Weight pack: fp32 [K,N] row-major -> bf16 fragment-major layout.
// dword idx = ((kt*(N/16) + n16)*32 + lane)*8 + v
//   lane<16 : packs (W[k][n], W[k+1][n]), k = kt*32 + 2v + (v>=4 ? 8 : 0)   (B-frag K=0..7,16..23)
//   lane>=16: same with k += 8                                              (B-frag K=8..15,24..31)
// => a wave's B fragment is 32 lanes x 32B contiguous = 2x global_load_b128.
// ---------------------------------------------------------------------------
__global__ __launch_bounds__(256)
void pack_w_kernel(const float* __restrict__ W, unsigned int* __restrict__ Wp,
                   int K, int N) {
  int idx = blockIdx.x * 256 + threadIdx.x;        // one dword per thread
  if (idx >= (K * N) / 2) return;
  int v    = idx & 7;
  int lane = (idx >> 3) & 31;
  int tile = idx >> 8;                              // kt*(N/16) + n16
  int nt   = N >> 4;
  int n16  = tile % nt;
  int kt   = tile / nt;
  int n    = (n16 << 4) + (lane & 15);
  int k0   = kt * 32 + ((lane >> 4) & 1) * 8 + 2 * v + (v >= 4 ? 8 : 0);
  unsigned int lo = f2bf(W[(size_t)k0 * N + n]);
  unsigned int hi = f2bf(W[(size_t)(k0 + 1) * N + n]);
  Wp[idx] = lo | (hi << 16);
}

// ---------------------------------------------------------------------------
// LayerNorm over 1024 cols: fp32 in -> bf16 out (row-major). 1 block per row.
// ---------------------------------------------------------------------------
__global__ __launch_bounds__(256)
void ln_bf16_kernel(const float* __restrict__ x, const float* __restrict__ g,
                    const float* __restrict__ b, unsigned short* __restrict__ out) {
  __shared__ float red[8];
  const int row = blockIdx.x;
  const int t   = threadIdx.x;
  const float4 v = ((const float4*)(x + (size_t)row * EMBED))[t];

  float s = v.x + v.y + v.z + v.w;
#pragma unroll
  for (int o = 16; o; o >>= 1) s += __shfl_xor(s, o, 32);
  if ((t & 31) == 0) red[t >> 5] = s;
  __syncthreads();
  float mu = 0.f;
#pragma unroll
  for (int i = 0; i < 8; i++) mu += red[i];
  mu *= (1.f / EMBED);
  __syncthreads();

  const float dx = v.x - mu, dy = v.y - mu, dz = v.z - mu, dw = v.w - mu;
  float s2 = dx * dx + dy * dy + dz * dz + dw * dw;
#pragma unroll
  for (int o = 16; o; o >>= 1) s2 += __shfl_xor(s2, o, 32);
  if ((t & 31) == 0) red[t >> 5] = s2;
  __syncthreads();
  float var = 0.f;
#pragma unroll
  for (int i = 0; i < 8; i++) var += red[i];
  var *= (1.f / EMBED);
  const float rs = rsqrtf(var + 1e-5f);

  const float4 gg = ((const float4*)g)[t];
  const float4 bb = ((const float4*)b)[t];
  const float y0 = dx * rs * gg.x + bb.x;
  const float y1 = dy * rs * gg.y + bb.y;
  const float y2 = dz * rs * gg.z + bb.z;
  const float y3 = dw * rs * gg.w + bb.w;
  uint2 pk;
  pk.x = (unsigned)f2bf(y0) | ((unsigned)f2bf(y1) << 16);
  pk.y = (unsigned)f2bf(y2) | ((unsigned)f2bf(y3) << 16);
  ((uint2*)(out + (size_t)row * EMBED))[t] = pk;
}

// ---------------------------------------------------------------------------
// bf16 GEMM: C[M,N] = A[M,K] (bf16 row-major) @ Wp (fragment-packed bf16)
// Wave tile 32x64 (2 A-frags x 4 B-frags -> 8 WMMA per K-step of 32).
// Block: 8 waves, 128x128 tile.   Modes:
//   0: out bf16 = C
//   1: out f32  = C + bias + resid
//   2: out bf16 = gelu_exact(C + bias)
// ---------------------------------------------------------------------------
__global__ __launch_bounds__(256)
void gemm_bf16_kernel(const unsigned short* __restrict__ A,
                      const unsigned int* __restrict__ Wp,
                      const float* __restrict__ bias,
                      const float* __restrict__ resid,
                      float* __restrict__ outF, unsigned short* __restrict__ outB,
                      int N, int kIters, int mode) {
  const int lane = threadIdx.x & 31;
  const int wid  = threadIdx.x >> 5;
  const int hi   = (lane >> 4) & 1;
  const int ln   = lane & 15;
  const int rowbase = blockIdx.x * 128 + (wid & 3) * 32;
  const int colbase = blockIdx.y * 128 + (wid >> 2) * 64;
  const int K = kIters * 32;
  const int ntiles = N >> 4;
  const int ntile0 = colbase >> 4;

  v8f acc[2][4];
#pragma unroll
  for (int i = 0; i < 2; i++)
#pragma unroll
    for (int j = 0; j < 4; j++) acc[i][j] = v8f{0.f,0.f,0.f,0.f,0.f,0.f,0.f,0.f};

  const unsigned short* a0p = A + (size_t)(rowbase + ln) * K;
  const unsigned short* a1p = A + (size_t)(rowbase + 16 + ln) * K;

  for (int kt = 0; kt < kIters; ++kt) {
    const int kc = kt * 32 + hi * 8;
    Frag a[2];
    a[0].q[0] = *(const uint4*)(a0p + kc);
    a[0].q[1] = *(const uint4*)(a0p + kc + 16);
    a[1].q[0] = *(const uint4*)(a1p + kc);
    a[1].q[1] = *(const uint4*)(a1p + kc + 16);
    Frag bfrag[4];
#pragma unroll
    for (int nf = 0; nf < 4; ++nf) {
      const uint4* bp = (const uint4*)(Wp +
          (((size_t)kt * ntiles + (ntile0 + nf)) * 32 + lane) * 8);
      bfrag[nf].q[0] = bp[0];
      bfrag[nf].q[1] = bp[1];
    }
#pragma unroll
    for (int mi = 0; mi < 2; ++mi)
#pragma unroll
      for (int nf = 0; nf < 4; ++nf)
        acc[mi][nf] = wmma_bf16(a[mi], bfrag[nf], acc[mi][nf]);
  }

#pragma unroll
  for (int mi = 0; mi < 2; ++mi) {
#pragma unroll
    for (int nf = 0; nf < 4; ++nf) {
      const size_t col = colbase + nf * 16 + ln;
      const float bs = bias ? bias[col] : 0.f;
#pragma unroll
      for (int r = 0; r < 8; ++r) {
        const size_t row = rowbase + mi * 16 + r + hi * 8;
        const size_t off = row * (size_t)N + col;
        const float v = acc[mi][nf][r];
        if (mode == 0) {
          outB[off] = f2bf(v);
        } else if (mode == 1) {
          outF[off] = v + bs + resid[off];
        } else {
          const float t = v + bs;
          outB[off] = f2bf(0.5f * t * (1.f + erff(t * 0.70710678118654752f)));
        }
      }
    }
  }
}

// ---------------------------------------------------------------------------
// V transpose: bf16 [M, E] row-major -> Vt bf16 [B, H, HD, T]
// ---------------------------------------------------------------------------
__global__ __launch_bounds__(256)
void vtrans_kernel(const unsigned short* __restrict__ V,
                   unsigned short* __restrict__ Vt) {
  const int idx = blockIdx.x * 256 + threadIdx.x;
  const int row = idx >> 10;            // token row in [0, 8192)
  const int c   = idx & 1023;           // col in [0, 1024)
  const int b = row >> 11, t = row & 2047;
  const int h = c >> 6,   d = c & 63;
  Vt[(((size_t)(b * NHEAD + h) * HD) + d) * SEQ + t] = V[(size_t)row * EMBED + c];
}

// ---------------------------------------------------------------------------
// Causal flash attention. One wave per 16 query rows of one (b,h).
// Q,K row-major bf16 [M,E]; Vt bf16 [B,H,HD,T]; out bf16 row-major [M,E].
// ---------------------------------------------------------------------------
__global__ __launch_bounds__(256)
void attn_kernel(const unsigned short* __restrict__ Q,
                 const unsigned short* __restrict__ Kb,
                 const unsigned short* __restrict__ Vt,
                 unsigned short* __restrict__ Ob) {
  __shared__ __align__(16) unsigned short Pbuf[8][16 * 32];  // per-wave P tile

  const int wid  = threadIdx.x >> 5;
  const int lane = threadIdx.x & 31;
  const int hi   = (lane >> 4) & 1;
  const int ln   = lane & 15;
  const int w  = blockIdx.x * 8 + wid;
  const int it = w & 127;                 // q tile within sequence
  const int bh = w >> 7;
  const int b = bh >> 4, h = bh & 15;
  const int i0 = it * 16;

  // Q fragments (16x64, split into two 16x32 A-frags), resident for whole loop
  Frag qf[2];
  const unsigned short* qrow = Q + (size_t)(b * SEQ + i0 + ln) * EMBED + h * HD;
#pragma unroll
  for (int ks = 0; ks < 2; ++ks) {
    qf[ks].q[0] = *(const uint4*)(qrow + ks * 32 + hi * 8);
    qf[ks].q[1] = *(const uint4*)(qrow + ks * 32 + hi * 8 + 16);
  }

  v8f O[4];
#pragma unroll
  for (int d = 0; d < 4; ++d) O[d] = v8f{0.f,0.f,0.f,0.f,0.f,0.f,0.f,0.f};
  float mrow[8], lrow[8];
#pragma unroll
  for (int r = 0; r < 8; ++r) { mrow[r] = -3.0e38f; lrow[r] = 0.f; }

  const float sc = 0.125f;                 // 1/sqrt(HD)
  const int nblk = (i0 + 16 + 31) >> 5;    // causal: 32-wide j blocks

  for (int jb = 0; jb < nblk; ++jb) {
    const int j0 = jb * 32;
    v8f s[2] = { v8f{0.f,0.f,0.f,0.f,0.f,0.f,0.f,0.f},
                 v8f{0.f,0.f,0.f,0.f,0.f,0.f,0.f,0.f} };
    // S = Q @ K^T : contraction over head-dim (contiguous in K's storage)
#pragma unroll
    for (int jt = 0; jt < 2; ++jt) {
      const unsigned short* kr =
          Kb + (size_t)(b * SEQ + j0 + jt * 16 + ln) * EMBED + h * HD + hi * 8;
#pragma unroll
      for (int ks = 0; ks < 2; ++ks) {
        Frag kf;
        kf.q[0] = *(const uint4*)(kr + ks * 32);
        kf.q[1] = *(const uint4*)(kr + ks * 32 + 16);
        s[jt] = wmma_bf16(qf[ks], kf, s[jt]);
      }
    }
    // online softmax, P -> LDS (bf16)
#pragma unroll
    for (int r = 0; r < 8; ++r) {
      const int irow = i0 + r + hi * 8;
      float a0 = s[0][r] * sc;
      float a1 = s[1][r] * sc;
      if (j0 + ln > irow)      a0 = -3.0e38f;
      if (j0 + 16 + ln > irow) a1 = -3.0e38f;
      float mx = fmaxf(a0, a1);
#pragma unroll
      for (int o = 8; o; o >>= 1) mx = fmaxf(mx, __shfl_xor(mx, o, 16));
      const float mnew = fmaxf(mrow[r], mx);
      const float corr = __expf(mrow[r] - mnew);
      const float p0 = __expf(a0 - mnew);
      const float p1 = __expf(a1 - mnew);
      float rsum = p0 + p1;
#pragma unroll
      for (int o = 8; o; o >>= 1) rsum += __shfl_xor(rsum, o, 16);
      lrow[r] = lrow[r] * corr + rsum;
      mrow[r] = mnew;
#pragma unroll
      for (int d = 0; d < 4; ++d) O[d][r] *= corr;
      unsigned short* pr = &Pbuf[wid][(r + hi * 8) * 32];
      pr[ln]      = f2bf(p0);
      pr[16 + ln] = f2bf(p1);
    }
    asm volatile("s_wait_dscnt 0" ::: "memory");
    // reload P as an A-fragment (16x32)
    Frag pa;
    const unsigned short* pl = &Pbuf[wid][ln * 32 + hi * 8];
    pa.q[0] = *(const uint4*)(pl);
    pa.q[1] = *(const uint4*)(pl + 16);
    // O += P @ V  (Vt gives contiguous B-frags along j)
#pragma unroll
    for (int d = 0; d < 4; ++d) {
      const unsigned short* vb = Vt +
          ((size_t)(b * NHEAD + h) * HD + d * 16 + ln) * SEQ + j0 + hi * 8;
      Frag vf;
      vf.q[0] = *(const uint4*)(vb);
      vf.q[1] = *(const uint4*)(vb + 16);
      O[d] = wmma_bf16(pa, vf, O[d]);
    }
  }

  // epilogue: O / l  -> bf16 row-major [M, E]
#pragma unroll
  for (int r = 0; r < 8; ++r) {
    const float inv = 1.f / lrow[r];
    const size_t row = (size_t)(b * SEQ + i0 + r + hi * 8);
#pragma unroll
    for (int d = 0; d < 4; ++d) {
      const size_t col = h * HD + d * 16 + ln;
      Ob[row * EMBED + col] = f2bf(O[d][r] * inv);
    }
  }
}

// ---------------------------------------------------------------------------
// Host launcher
// ---------------------------------------------------------------------------
extern "C" void kernel_launch(void* const* d_in, const int* in_sizes, int n_in,
                              void* d_out, int out_size, void* d_ws, size_t ws_size,
                              hipStream_t stream) {
  (void)in_sizes; (void)n_in; (void)out_size; (void)ws_size;
  const float* x   = (const float*)d_in[0];
  const float* Wq  = (const float*)d_in[1];
  const float* Wk  = (const float*)d_in[2];
  const float* Wv  = (const float*)d_in[3];
  const float* Wo  = (const float*)d_in[4];
  const float* bo  = (const float*)d_in[5];
  const float* W1  = (const float*)d_in[6];
  const float* b1  = (const float*)d_in[7];
  const float* W2  = (const float*)d_in[8];
  const float* b2  = (const float*)d_in[9];
  const float* g1  = (const float*)d_in[10];
  const float* be1 = (const float*)d_in[11];
  const float* g2  = (const float*)d_in[12];
  const float* be2 = (const float*)d_in[13];
  float* out = (float*)d_out;

  char* ws = (char*)d_ws;
  size_t off = 0;
  auto alloc = [&](size_t bytes) -> void* {
    void* p = ws + off;
    off += (bytes + 255) & ~(size_t)255;
    return p;
  };
  unsigned int*   WqP = (unsigned int*)alloc((size_t)EMBED * EMBED * 2);
  unsigned int*   WkP = (unsigned int*)alloc((size_t)EMBED * EMBED * 2);
  unsigned int*   WvP = (unsigned int*)alloc((size_t)EMBED * EMBED * 2);
  unsigned int*   WoP = (unsigned int*)alloc((size_t)EMBED * EMBED * 2);
  unsigned int*   W1P = (unsigned int*)alloc((size_t)EMBED * FFDIM * 2);
  unsigned int*   W2P = (unsigned int*)alloc((size_t)FFDIM * EMBED * 2);
  unsigned short* h1  = (unsigned short*)alloc((size_t)MTOT * EMBED * 2);
  unsigned short* Qb  = (unsigned short*)alloc((size_t)MTOT * EMBED * 2);
  unsigned short* Kbm = (unsigned short*)alloc((size_t)MTOT * EMBED * 2);
  unsigned short* Vb  = (unsigned short*)alloc((size_t)MTOT * EMBED * 2);
  unsigned short* VtB = (unsigned short*)alloc((size_t)MTOT * EMBED * 2);
  unsigned short* aO  = (unsigned short*)alloc((size_t)MTOT * EMBED * 2);
  float*          xm  = (float*)alloc((size_t)MTOT * EMBED * 4);
  unsigned short* h2  = (unsigned short*)alloc((size_t)MTOT * EMBED * 2);
  unsigned short* a1  = (unsigned short*)alloc((size_t)MTOT * FFDIM * 2);

  // 1) pack weights fp32 -> fragment-packed bf16
  pack_w_kernel<<<(EMBED * EMBED / 2) / 256, 256, 0, stream>>>(Wq, WqP, EMBED, EMBED);
  pack_w_kernel<<<(EMBED * EMBED / 2) / 256, 256, 0, stream>>>(Wk, WkP, EMBED, EMBED);
  pack_w_kernel<<<(EMBED * EMBED / 2) / 256, 256, 0, stream>>>(Wv, WvP, EMBED, EMBED);
  pack_w_kernel<<<(EMBED * EMBED / 2) / 256, 256, 0, stream>>>(Wo, WoP, EMBED, EMBED);
  pack_w_kernel<<<(EMBED * FFDIM / 2) / 256, 256, 0, stream>>>(W1, W1P, EMBED, FFDIM);
  pack_w_kernel<<<(FFDIM * EMBED / 2) / 256, 256, 0, stream>>>(W2, W2P, FFDIM, EMBED);

  // 2) LN1: x -> h1 (bf16)
  ln_bf16_kernel<<<MTOT, 256, 0, stream>>>(x, g1, be1, h1);

  // 3) Q/K/V projections (no bias in reference)
  dim3 gE(MTOT / 128, EMBED / 128);
  gemm_bf16_kernel<<<gE, 256, 0, stream>>>(h1, WqP, nullptr, nullptr, nullptr, Qb,  EMBED, EMBED / 32, 0);
  gemm_bf16_kernel<<<gE, 256, 0, stream>>>(h1, WkP, nullptr, nullptr, nullptr, Kbm, EMBED, EMBED / 32, 0);
  gemm_bf16_kernel<<<gE, 256, 0, stream>>>(h1, WvP, nullptr, nullptr, nullptr, Vb,  EMBED, EMBED / 32, 0);

  // 4) V -> Vt [B,H,HD,T]
  vtrans_kernel<<<(MTOT * EMBED) / 256, 256, 0, stream>>>(Vb, VtB);

  // 5) causal flash attention -> aO (bf16 row-major)
  attn_kernel<<<(BATCH * NHEAD * (SEQ / 16)) / 8, 256, 0, stream>>>(Qb, Kbm, VtB, aO);

  // 6) output projection + bo + residual(x) -> xm (fp32)
  gemm_bf16_kernel<<<gE, 256, 0, stream>>>(aO, WoP, bo, x, xm, nullptr, EMBED, EMBED / 32, 1);

  // 7) LN2: xm -> h2 (bf16)
  ln_bf16_kernel<<<MTOT, 256, 0, stream>>>(xm, g2, be2, h2);

  // 8) FFN: gelu(h2@W1 + b1) -> a1 (bf16)
  dim3 gF1(MTOT / 128, FFDIM / 128);
  gemm_bf16_kernel<<<gF1, 256, 0, stream>>>(h2, W1P, b1, nullptr, nullptr, a1, FFDIM, EMBED / 32, 2);

  // 9) a1@W2 + b2 + residual(xm) -> out (fp32)
  gemm_bf16_kernel<<<gE, 256, 0, stream>>>(a1, W2P, b2, xm, out, nullptr, EMBED, FFDIM / 32, 1);
}
